// SelfAttention_73718818669277
// MI455X (gfx1250) — matrix operand order repaired and checked by reading, not compiled, not run on hardware
//
#include <hip/hip_runtime.h>
#include <hip/hip_bf16.h>
#include <math.h>

typedef __attribute__((ext_vector_type(16))) _Float16 v16h;
typedef __attribute__((ext_vector_type(8)))  float    v8f;

#define CC 256
#define NN_TOT 2048
#define BB 8
#define NTILES 128   // NN_TOT/16
#define JBLK 64      // NN_TOT/32

// Packed fragment layouts (all f16, 32B per lane per fragment, coalesced):
//  WP : [mat(4)][cotile(16)][ck(8)][lane(32)][h(16)]   A-frags of weights
//  QP : [b][ntile(128)][ck(8)][lane][h]                A-frags of Q^T (M=n,K=c)
//  KP : [b][mtile(128)][ck(8)][lane][h]                B-frags of K   (K=c,N=m)
//  VP : [b][ck16(16)][j(64)][lane][h]                  A-frags of V   (M=c,K=m)
//  AP : [b][ntile(128)][ck(8)][lane][h]                B-frags of av  (K=c,N=n)

__device__ __forceinline__ v16h ld16(const _Float16* p) { return *(const v16h*)p; }
__device__ __forceinline__ void st16(_Float16* p, v16h v) { *(v16h*)p = v; }

// async 16B global->LDS copy, tracked by ASYNCcnt (CDNA5)
__device__ __forceinline__ void async_cp16(unsigned lds_off, const _Float16* g) {
  asm volatile("global_load_async_to_lds_b128 %0, %1, off"
               :: "v"(lds_off), "v"(g) : "memory");
}
__device__ __forceinline__ void wait_async_le32() {
  asm volatile("s_wait_asynccnt 0x20" ::: "memory");
}
__device__ __forceinline__ void wait_async_0() {
  asm volatile("s_wait_asynccnt 0x0" ::: "memory");
}

// ---- kernel 0: pack weights into f16 A-fragment order -----------------------
__global__ void __launch_bounds__(256) k_pack_w(const float* __restrict__ wq,
                                                const float* __restrict__ wk,
                                                const float* __restrict__ wv,
                                                const float* __restrict__ wo,
                                                _Float16* __restrict__ WP) {
  int id = blockIdx.x * 256 + threadIdx.x;      // 4*16*8*32*16 = 262144 halves
  int h      = id & 15;
  int lane   = (id >> 4) & 31;
  int ck     = (id >> 9) & 7;
  int cotile = (id >> 12) & 15;
  int mat    = (id >> 16) & 3;
  int k  = ck * 32 + ((lane & 16) ? 16 : 0) + h;
  int co = cotile * 16 + (lane & 15);
  const float* w = (mat == 0) ? wq : (mat == 1) ? wk : (mat == 2) ? wv : wo;
  WP[id] = (_Float16)w[co * CC + k];
}

// ---- kernel 1: Q/K/V projections, outputs in packed fragment order ----------
__global__ void __launch_bounds__(256) k_proj_qkv(
    const float* __restrict__ x, const _Float16* __restrict__ WP,
    const float* __restrict__ bq, const float* __restrict__ bk,
    const float* __restrict__ bv,
    _Float16* __restrict__ QP, _Float16* __restrict__ KP,
    _Float16* __restrict__ VP) {
  const int b  = blockIdx.y;
  const int nt = blockIdx.x;          // 16-column tile index
  const int n0 = nt * 16;
  __shared__ __align__(32) _Float16 xT[16 * CC];  // [n][ci] f16, 8KB

  for (int idx = threadIdx.x; idx < 16 * CC; idx += 256) {
    int ci = idx >> 4, n = idx & 15;
    xT[n * CC + ci] = (_Float16)x[((size_t)(b * CC + ci)) * NN_TOT + n0 + n];
  }
  __syncthreads();

  const int wave = threadIdx.x >> 5, lane = threadIdx.x & 31;
  const int nn      = lane & 15;
  const int rowBase = (lane >> 4) << 3;

  for (int t = wave; t < 48; t += 8) {          // 3 matrices x 16 co-tiles
    const int mat = t >> 4, cotile = t & 15;
    const float* bb = (mat == 0) ? bq : (mat == 1) ? bk : bv;

    v16h a[8];
#pragma unroll
    for (int ck = 0; ck < 8; ++ck)
      a[ck] = ld16(WP + ((((size_t)mat * 16 + cotile) * 8 + ck) * 32 + lane) * 16);

    v8f d0, d1;
#pragma unroll
    for (int i = 0; i < 8; ++i) { d0[i] = 0.f; d1[i] = 0.f; }
#pragma unroll
    for (int ck = 0; ck < 8; ck += 2) {
      v16h bf0 = ld16(&xT[nn * CC + ck * 32 + ((lane & 16) ? 16 : 0)]);
      v16h bf1 = ld16(&xT[nn * CC + (ck + 1) * 32 + ((lane & 16) ? 16 : 0)]);
      d0 = __builtin_amdgcn_wmma_f32_16x16x32_f16(false, a[ck],     false, bf0, (short)0, d0, false, false);
      d1 = __builtin_amdgcn_wmma_f32_16x16x32_f16(false, a[ck + 1], false, bf1, (short)0, d1, false, false);
    }
#pragma unroll
    for (int i = 0; i < 8; ++i) {
      int co = cotile * 16 + rowBase + i;
      _Float16 val = (_Float16)(d0[i] + d1[i] + bb[co]);
      if (mat == 0) {          // Q as A-frag: M=n, K=c
        int Ld = nn + (((co >> 4) & 1) << 4);
        int hh = co & 15;
        QP[((((size_t)b * NTILES + nt) * 8 + (co >> 5)) * 32 + Ld) * 16 + hh] = val;
      } else if (mat == 1) {   // K as B-frag: K=c, N=m(=n here)
        int Ld = nn + (((co >> 4) & 1) << 4);
        KP[((((size_t)b * NTILES + nt) * 8 + (co >> 5)) * 32 + Ld) * 16 + (co & 15)] = val;
      } else {                 // V as A-frag: M=c, K=m(=n here)
        int m = n0 + nn, mlow = m & 31;
        int Ld = (co & 15) + ((mlow & 16) ? 16 : 0);
        int hh = mlow & 15;
        VP[((((size_t)b * 16 + (co >> 4)) * JBLK + (m >> 5)) * 32 + Ld) * 16 + hh] = val;
      }
    }
  }
}

// stage one 16KB V key-block (16 fragments) into LDS via async copies (32 ops)
__device__ __forceinline__ void stage_V(const _Float16* __restrict__ VP, int b,
                                        int j, int lane, unsigned ldsBase) {
#pragma unroll
  for (int t = 0; t < 16; ++t) {
    const _Float16* g = VP + ((((size_t)b * 16 + t) * JBLK + j) * 32 + lane) * 16;
    unsigned d = ldsBase + (unsigned)((t * 32 + lane) * 32);
    async_cp16(d, g);          // lane's first 16B
    async_cp16(d + 16, g + 8); // lane's second 16B
  }
}

// ---- kernel 2: flash attention (one wave per 16-query tile) -----------------
__global__ void __launch_bounds__(32) k_attn(const _Float16* __restrict__ QP,
                                             const _Float16* __restrict__ KP,
                                             const _Float16* __restrict__ VP,
                                             _Float16* __restrict__ AP) {
  const int b  = blockIdx.y;
  const int nt = blockIdx.x;
  const int lane    = threadIdx.x;
  const int nn      = lane & 15;
  const int rowBase = (lane >> 4) << 3;

  __shared__ __align__(32) _Float16 ldsQ[8 * 32 * 16];       // 8KB packed A-frags
  __shared__ __align__(32) _Float16 ldsV[2][16 * 32 * 16];   // 2 x 16KB V blocks
  __shared__ __align__(32) _Float16 ldsP[16 * 32];           // P^T staging
  __shared__ float ldsStat[16];

  const unsigned vbase0 = (unsigned)(unsigned long long)&ldsV[0][0];
  const unsigned vbase1 = (unsigned)(unsigned long long)&ldsV[1][0];

  // kick off first V block asynchronously, then stage Q into LDS
  stage_V(VP, b, 0, lane, vbase0);
#pragma unroll
  for (int ck = 0; ck < 8; ++ck)
    st16(&ldsQ[(ck * 32 + lane) * 16],
         ld16(QP + ((((size_t)b * NTILES + nt) * 8 + ck) * 32 + lane) * 16));
  __syncthreads();

  v8f acc[16];
#pragma unroll
  for (int t = 0; t < 16; ++t)
#pragma unroll
    for (int i = 0; i < 8; ++i) acc[t][i] = 0.f;

  float m_run[8], l_run[8];
#pragma unroll
  for (int i = 0; i < 8; ++i) { m_run[i] = -3.0e38f; l_run[i] = 0.f; }

  const float norm = 0.0625f;  // 1/sqrt(256)

  for (int j = 0; j < JBLK; ++j) {
    v8f s0a, s0b, s1a, s1b;    // 4 independent WMMA chains
#pragma unroll
    for (int i = 0; i < 8; ++i) { s0a[i] = 0.f; s0b[i] = 0.f; s1a[i] = 0.f; s1b[i] = 0.f; }

#pragma unroll
    for (int ck = 0; ck < 8; ck += 2) {
      v16h aq0 = ld16(&ldsQ[(ck * 32 + lane) * 16]);
      v16h aq1 = ld16(&ldsQ[((ck + 1) * 32 + lane) * 16]);
      const size_t kb0 = (((size_t)b * NTILES + 2 * j) * 8) * 32 * 16;
      const size_t kb1 = (((size_t)b * NTILES + 2 * j + 1) * 8) * 32 * 16;
      v16h bk00 = ld16(KP + kb0 + (((size_t)ck * 32 + lane) * 16));
      v16h bk01 = ld16(KP + kb0 + ((((size_t)ck + 1) * 32 + lane) * 16));
      v16h bk10 = ld16(KP + kb1 + (((size_t)ck * 32 + lane) * 16));
      v16h bk11 = ld16(KP + kb1 + ((((size_t)ck + 1) * 32 + lane) * 16));
      s0a = __builtin_amdgcn_wmma_f32_16x16x32_f16(false, aq0, false, bk00, (short)0, s0a, false, false);
      s1a = __builtin_amdgcn_wmma_f32_16x16x32_f16(false, aq0, false, bk10, (short)0, s1a, false, false);
      s0b = __builtin_amdgcn_wmma_f32_16x16x32_f16(false, aq1, false, bk01, (short)0, s0b, false, false);
      s1b = __builtin_amdgcn_wmma_f32_16x16x32_f16(false, aq1, false, bk11, (short)0, s1b, false, false);
    }

    // overlap: issue next V block async copies + prefetch next K block
    if (j + 1 < JBLK) {
      stage_V(VP, b, j + 1, lane, (j & 1) ? vbase0 : vbase1);
      const char* kpn = (const char*)(KP + (((size_t)b * NTILES + 2 * (j + 1)) * 8) * 32 * 16);
      __builtin_prefetch(kpn + lane * 512 + 0,   0, 1);
      __builtin_prefetch(kpn + lane * 512 + 128, 0, 1);
      __builtin_prefetch(kpn + lane * 512 + 256, 0, 1);
      __builtin_prefetch(kpn + lane * 512 + 384, 0, 1);
      wait_async_le32();     // current block's 32 copies have landed
    } else {
      wait_async_0();
    }

    float e0[8], e1[8], al[8];
#pragma unroll
    for (int i = 0; i < 8; ++i) {
      float a0 = (s0a[i] + s0b[i]) * norm;
      float a1 = (s1a[i] + s1b[i]) * norm;
      float t = fmaxf(a0, a1);
      t = fmaxf(t, __shfl_xor(t, 1, 16));
      t = fmaxf(t, __shfl_xor(t, 2, 16));
      t = fmaxf(t, __shfl_xor(t, 4, 16));
      t = fmaxf(t, __shfl_xor(t, 8, 16));
      float nm = fmaxf(m_run[i], t);
      al[i]   = __expf(m_run[i] - nm);
      m_run[i] = nm;
      e0[i] = __expf(a0 - nm);
      e1[i] = __expf(a1 - nm);
      float rs = e0[i] + e1[i];
      rs += __shfl_xor(rs, 1, 16);
      rs += __shfl_xor(rs, 2, 16);
      rs += __shfl_xor(rs, 4, 16);
      rs += __shfl_xor(rs, 8, 16);
      l_run[i] = l_run[i] * al[i] + rs;
    }

    if (nn == 0) {   // publish per-row alpha (row-indexed -> lane-indexed)
#pragma unroll
      for (int i = 0; i < 8; ++i) ldsStat[rowBase + i] = al[i];
    }
#pragma unroll
    for (int i = 0; i < 8; ++i) {  // stage P^T tile for the PV B-fragment
      int row = rowBase + i;
      ldsP[row * 32 + nn]      = (_Float16)e0[i];
      ldsP[row * 32 + 16 + nn] = (_Float16)e1[i];
    }
    __syncthreads();

    float af = ldsStat[nn];
    v16h pf  = ld16(&ldsP[nn * 32 + ((lane & 16) ? 16 : 0)]);
    const _Float16* vcur = (j & 1) ? &ldsV[1][0] : &ldsV[0][0];
#pragma unroll
    for (int t = 0; t < 16; ++t) {
#pragma unroll
      for (int i = 0; i < 8; ++i) acc[t][i] *= af;
      v16h av = ld16(vcur + (t * 32 + lane) * 16);
      acc[t] = __builtin_amdgcn_wmma_f32_16x16x32_f16(false, av, false, pf,
                                                      (short)0, acc[t], false, false);
    }
    __syncthreads();
  }

  if (nn == 0) {
#pragma unroll
    for (int i = 0; i < 8; ++i) ldsStat[rowBase + i] = l_run[i];
  }
  __syncthreads();
  float inv = 1.0f / ldsStat[nn];
#pragma unroll
  for (int t = 0; t < 16; ++t) {
#pragma unroll
    for (int i = 0; i < 8; ++i) {
      int c = t * 16 + rowBase + i;                 // channel
      int Ld = nn + (((c >> 4) & 1) << 4);          // av as B-frag: K=c, N=n
      AP[((((size_t)b * NTILES + nt) * 8 + (c >> 5)) * 32 + Ld) * 16 + (c & 15)] =
          (_Float16)(acc[t][i] * inv);
    }
  }
}

// ---- kernel 3: output projection + residual ---------------------------------
__global__ void __launch_bounds__(256) k_proj_o(const float* __restrict__ x,
                                                const _Float16* __restrict__ WP,
                                                const float* __restrict__ bo,
                                                const _Float16* __restrict__ AP,
                                                float* __restrict__ out) {
  const int b  = blockIdx.y;
  const int nt = blockIdx.x;
  const int n0 = nt * 16;
  const int wave = threadIdx.x >> 5, lane = threadIdx.x & 31;
  const int nn      = lane & 15;
  const int rowBase = (lane >> 4) << 3;

  for (int cotile = wave; cotile < 16; cotile += 8) {
    v16h a[8], bf[8];
#pragma unroll
    for (int ck = 0; ck < 8; ++ck) {
      a[ck]  = ld16(WP + ((((size_t)3 * 16 + cotile) * 8 + ck) * 32 + lane) * 16);
      bf[ck] = ld16(AP + ((((size_t)b * NTILES + nt) * 8 + ck) * 32 + lane) * 16);
    }
    v8f d0, d1;
#pragma unroll
    for (int i = 0; i < 8; ++i) { d0[i] = 0.f; d1[i] = 0.f; }
#pragma unroll
    for (int ck = 0; ck < 8; ck += 2) {
      d0 = __builtin_amdgcn_wmma_f32_16x16x32_f16(false, a[ck],     false, bf[ck],     (short)0, d0, false, false);
      d1 = __builtin_amdgcn_wmma_f32_16x16x32_f16(false, a[ck + 1], false, bf[ck + 1], (short)0, d1, false, false);
    }
#pragma unroll
    for (int i = 0; i < 8; ++i) {
      int co = cotile * 16 + rowBase + i;
      size_t idx = ((size_t)(b * CC + co)) * NN_TOT + n0 + nn;
      out[idx] = x[idx] + d0[i] + d1[i] + bo[co];
    }
  }
}

// ---- launcher ---------------------------------------------------------------
extern "C" void kernel_launch(void* const* d_in, const int* in_sizes, int n_in,
                              void* d_out, int out_size, void* d_ws, size_t ws_size,
                              hipStream_t stream) {
  (void)in_sizes; (void)n_in; (void)out_size;
  const float* x  = (const float*)d_in[0];
  const float* wq = (const float*)d_in[1];
  const float* bq = (const float*)d_in[2];
  const float* wk = (const float*)d_in[3];
  const float* bk = (const float*)d_in[4];
  const float* wv = (const float*)d_in[5];
  const float* bv = (const float*)d_in[6];
  const float* wo = (const float*)d_in[7];
  const float* bo = (const float*)d_in[8];
  float* out = (float*)d_out;

  const size_t WPB = (size_t)4 * 16 * 8 * 32 * 16 * 2;          // 512 KB
  const size_t FRB = (size_t)BB * NTILES * 8 * 32 * 16 * 2;     // 8 MB each
  if (ws_size < WPB + 4 * FRB) return;                          // ~32.5 MB
  char* ws = (char*)d_ws;
  _Float16* WP = (_Float16*)(ws);
  _Float16* QP = (_Float16*)(ws + WPB);
  _Float16* KP = (_Float16*)(ws + WPB + FRB);
  _Float16* VP = (_Float16*)(ws + WPB + 2 * FRB);
  _Float16* AP = (_Float16*)(ws + WPB + 3 * FRB);

  dim3 grid(NTILES, BB);
  k_pack_w<<<1024, 256, 0, stream>>>(wq, wk, wv, wo, WP);
  k_proj_qkv<<<grid, 256, 0, stream>>>(x, WP, bq, bk, bv, QP, KP, VP);
  k_attn<<<grid, 32, 0, stream>>>(QP, KP, VP, AP);
  k_proj_o<<<grid, 256, 0, stream>>>(x, WP, bo, AP, out);
}
